// SyntaxTreeNetwork_82136954569227
// MI455X (gfx1250) — matrix operand order repaired and checked by reading, not compiled, not run on hardware
//
#include <hip/hip_runtime.h>
#include <hip/hip_bf16.h>

#define Bdim 64
#define Ldim 512
#define Hdim 256
#define H2   512   // 2*H
#define Rdim 64
#define NTdim 32

typedef __attribute__((ext_vector_type(16))) _Float16 v16h;
typedef __attribute__((ext_vector_type(8)))  _Float16 v8h;
typedef __attribute__((ext_vector_type(8)))  float    v8f;

// ---------------------------------------------------------------------------
// tanh: use the gfx1250 hardware transcendental V_TANH_F32 when available,
// else a branch-free v_exp_f32 + v_rcp_f32 sequence.  Host pass sees tanhf.
// ---------------------------------------------------------------------------
__device__ __forceinline__ float htanh(float x) {
#if defined(__HIP_DEVICE_COMPILE__)
#if __has_builtin(__builtin_amdgcn_tanhf)
    return __builtin_amdgcn_tanhf(x);
#else
    // tanh(x) = copysign(1 - 2/(e^{2|x|}+1), x);  e^{2|x|} = 2^{2*log2(e)*|x|}
    float ax = __builtin_fabsf(x);
    float e  = __builtin_amdgcn_exp2f(2.8853900817779268f * ax);
    float r  = 1.0f - 2.0f * __builtin_amdgcn_rcpf(e + 1.0f);
    return __builtin_copysignf(r, x);
#endif
#else
    return tanhf(x);
#endif
}

// ---------------------------------------------------------------------------
// Fragment loaders for CDNA5 WMMA f16 16x16x32 (wave32); A and B use the same contiguous k order
// ---------------------------------------------------------------------------

// A (16x32, MxK), row-major in memory with row stride ldRow (halves).
// Lane L: m = L%16, hi = L/16.  elems 0..7 -> K = hi*16 + 0..7
//                               elems 8..15 -> K = hi*16 + 8..15
__device__ __forceinline__ v16h load_A_frag(const _Float16* __restrict__ tile,
                                            int ldRow, int lane) {
    const int m  = lane & 15;
    const int hi = lane >> 4;
    const _Float16* p = tile + (size_t)m * ldRow + hi * 16;
    union { v16h v; v8h h[2]; } u;
    u.h[0] = *(const v8h*)(p);
    u.h[1] = *(const v8h*)(p + 8);
    return u.v;
}

// B (32x16, KxN) where B[k][n] = W[n][k]; W row-major with row stride ldRow.
// Lane L: n = L%16, hi = L/16.  elems 0..15 -> K = hi*16 + 0..15 (contiguous)
__device__ __forceinline__ v16h load_B_frag(const _Float16* __restrict__ wrow0,
                                            int ldRow, int lane) {
    const int n  = lane & 15;
    const int hi = lane >> 4;
    return *(const v16h*)(wrow0 + (size_t)n * ldRow + hi * 16);
}

// ---------------------------------------------------------------------------
// Kernel 1: convert rule_weights f32 -> f16
// ---------------------------------------------------------------------------
__global__ __launch_bounds__(256) void convert_weights_kernel(
    const float* __restrict__ w32, _Float16* __restrict__ w16, int total) {
    int idx = blockIdx.x * 256 + threadIdx.x;
    if (idx < total) w16[idx] = (_Float16)w32[idx];
}

// ---------------------------------------------------------------------------
// Kernel 2: leaves  h = tanh(emb[token] + bias[preterm])  (store f16)
// ---------------------------------------------------------------------------
__global__ __launch_bounds__(256) void leaf_kernel(
    const int* __restrict__ token_ids, const int* __restrict__ preterm_nt,
    const float* __restrict__ emb, const float* __restrict__ biases,
    _Float16* __restrict__ hout, int total) {
    int idx = blockIdx.x * 256 + threadIdx.x;
    if (idx >= total) return;
    int hc = idx & (Hdim - 1);
    int bl = idx >> 8;                 // H == 256
    int t  = token_ids[bl];
    int p  = preterm_nt[bl];
    float v = htanh(emb[(size_t)t * Hdim + hc] + biases[(size_t)p * Hdim + hc]);
    hout[idx] = (_Float16)v;
}

// ---------------------------------------------------------------------------
// Kernel 3: one tree level.  One workgroup (8 wave32) per node.
// Each node: out(64x256) = tanh( h_agr(64x512) * W^T + bias )
// Wave w: M-tiles {2*(w&1), 2*(w&1)+1}, N-tiles [4*(w>>1) .. +4)
// ---------------------------------------------------------------------------
__global__ __launch_bounds__(256) void tree_level_kernel(
    const _Float16* __restrict__ hin,   // (B, 2*n_nodes, H) f16
    _Float16* __restrict__ hout,        // (B, n_nodes, H) f16
    const _Float16* __restrict__ W16,   // (R, H, 2H) f16
    const int* __restrict__ rule_ids, const int* __restrict__ nt_ids,
    const float* __restrict__ biases,   // (NT, H) f32
    int n_nodes, int offset,
    float* __restrict__ final_out)      // non-null only when n_nodes == 1
{
    const int node = blockIdx.x;
    const int tid  = threadIdx.x;
    const int lane = tid & 31;
    const int wave = tid >> 5;
    const int mgrp = wave & 1;   // 2 groups of 2 M-tiles
    const int ngrp = wave >> 1;  // 4 groups of 4 N-tiles

    const int r    = rule_ids[offset + node];
    const int ntid = nt_ids[offset + node];

    const _Float16* Wr    = W16 + (size_t)r * (Hdim * H2);
    const int       ldA   = 2 * n_nodes * Hdim;            // input batch-row stride
    const _Float16* Abase = hin + (size_t)node * H2;       // (m=0, k=0) of this node

    v8f acc[2][4] = {};

    for (int ks = 0; ks < 16; ++ks) {
        const int kbase = ks * 32;
        v16h a0 = load_A_frag(Abase + (size_t)(mgrp * 32 +  0) * ldA + kbase, ldA, lane);
        v16h a1 = load_A_frag(Abase + (size_t)(mgrp * 32 + 16) * ldA + kbase, ldA, lane);
        v16h b[4];
#pragma unroll
        for (int j = 0; j < 4; ++j)
            b[j] = load_B_frag(Wr + (size_t)(ngrp * 64 + j * 16) * H2 + kbase, H2, lane);
#pragma unroll
        for (int j = 0; j < 4; ++j) {
            acc[0][j] = __builtin_amdgcn_wmma_f32_16x16x32_f16(
                false, a0, false, b[j], (short)0, acc[0][j], false, false);
            acc[1][j] = __builtin_amdgcn_wmma_f32_16x16x32_f16(
                false, a1, false, b[j], (short)0, acc[1][j], false, false);
        }
    }

    // Epilogue: bias + tanh, store f16 (and f32 at the root level)
    const int n  = lane & 15;
    const int hi = lane >> 4;
    const int ldOut = n_nodes * Hdim;
#pragma unroll
    for (int mi = 0; mi < 2; ++mi) {
        const int mtile = mgrp * 2 + mi;
#pragma unroll
        for (int j = 0; j < 4; ++j) {
            const int col  = ngrp * 64 + j * 16 + n;
            const float bv = biases[(size_t)ntid * Hdim + col];
#pragma unroll
            for (int e = 0; e < 8; ++e) {
                const int br = mtile * 16 + hi * 8 + e;   // batch row
                const float v = htanh(acc[mi][j][e] + bv);
                hout[(size_t)br * ldOut + (size_t)node * Hdim + col] = (_Float16)v;
                if (final_out) final_out[(size_t)br * Hdim + col] = v;
            }
        }
    }
}

// ---------------------------------------------------------------------------
// Host launcher
// ---------------------------------------------------------------------------
extern "C" void kernel_launch(void* const* d_in, const int* in_sizes, int n_in,
                              void* d_out, int out_size, void* d_ws, size_t ws_size,
                              hipStream_t stream) {
    const int*   token_ids  = (const int*)d_in[0];
    const int*   preterm_nt = (const int*)d_in[1];
    const int*   rule_ids   = (const int*)d_in[2];
    const int*   nt_ids     = (const int*)d_in[3];
    const float* embedding  = (const float*)d_in[4];
    const float* rule_w     = (const float*)d_in[5];
    const float* biases     = (const float*)d_in[6];

    char* ws = (char*)d_ws;
    const size_t W16_BYTES  = (size_t)Rdim * Hdim * H2 * sizeof(_Float16);   // 16 MiB
    const size_t BUF0_BYTES = (size_t)Bdim * Ldim * Hdim * sizeof(_Float16); // 16 MiB
    _Float16* W16  = (_Float16*)(ws);
    _Float16* buf0 = (_Float16*)(ws + W16_BYTES);
    _Float16* buf1 = (_Float16*)(ws + W16_BYTES + BUF0_BYTES);               // <= 8 MiB used

    // 1) weights f32 -> f16
    {
        const int total = Rdim * Hdim * H2;                   // 8,388,608
        convert_weights_kernel<<<total / 256, 256, 0, stream>>>(rule_w, W16, total);
    }
    // 2) leaves
    {
        const int total = Bdim * Ldim * Hdim;                 // 8,388,608
        leaf_kernel<<<total / 256, 256, 0, stream>>>(token_ids, preterm_nt,
                                                     embedding, biases, buf0, total);
    }
    // 3) tree levels n = 256 .. 1 (ping-pong f16 buffers)
    int offset = 0;
    const _Float16* cur = buf0;
    _Float16*       nxt = buf1;
    for (int n = Ldim / 2; n >= 1; n >>= 1) {
        tree_level_kernel<<<n, 256, 0, stream>>>(
            cur, nxt, W16, rule_ids, nt_ids, biases, n, offset,
            (n == 1) ? (float*)d_out : nullptr);
        offset += n;
        const _Float16* t = cur; cur = nxt; nxt = (_Float16*)t;
    }
}
